// LSTM_35536559407740
// MI455X (gfx1250) — matrix operand order repaired and checked by reading, not compiled, not run on hardware
//
#include <hip/hip_runtime.h>
#include <hip/hip_bf16.h>

// Problem dims (fixed by reference): B=512, S=256, H=512, O=128
#define BD 512
#define SD 256
#define HD 512
#define OD 128
#define G4 2048   // 4*HD (gate-concatenated width, order: g,i,f,o)

typedef __attribute__((ext_vector_type(16))) __bf16 v16bf;
typedef __attribute__((ext_vector_type(8)))  __bf16 v8bf;
typedef __attribute__((ext_vector_type(8)))  float  v8f;

union Frag16 { v16bf v; v8bf h[2]; };

__device__ __forceinline__ float sigf(float x) {
    return 1.0f / (1.0f + __expf(-x));
}
__device__ __forceinline__ float tanhfast(float x) {
    float e = __expf(2.0f * x);
    return (e - 1.0f) / (e + 1.0f);
}

// ---------------------------------------------------------------------------
// Prep: fuse 4 hidden weight matrices into bf16 Wh (HD x 4H, row-major),
// and build Wx4[4H], bb4[4H] = b*x + b*h.
// ---------------------------------------------------------------------------
__global__ void lstm_prep(const float* __restrict__ Wgh, const float* __restrict__ Wih,
                          const float* __restrict__ Wfh, const float* __restrict__ Woh,
                          const float* __restrict__ Wgx, const float* __restrict__ Wix,
                          const float* __restrict__ Wfx, const float* __restrict__ Wox,
                          const float* __restrict__ bgx, const float* __restrict__ bix,
                          const float* __restrict__ bfx, const float* __restrict__ box,
                          const float* __restrict__ bgh, const float* __restrict__ bih,
                          const float* __restrict__ bfh, const float* __restrict__ boh,
                          __bf16* __restrict__ Whbf, float* __restrict__ Wx4,
                          float* __restrict__ bb4) {
    int idx = blockIdx.x * blockDim.x + threadIdx.x;
    if (idx >= HD * G4) return;
    int k   = idx >> 11;         // row (0..511)
    int col = idx & (G4 - 1);    // gate-concat column
    int g   = col >> 9;          // gate id: 0=g,1=i,2=f,3=o
    int j   = col & (HD - 1);
    const float* W = (g == 0) ? Wgh : (g == 1) ? Wih : (g == 2) ? Wfh : Woh;
    Whbf[idx] = (__bf16)W[k * HD + j];
    if (k == 0) {
        const float* wx = (g == 0) ? Wgx : (g == 1) ? Wix : (g == 2) ? Wfx : Wox;
        const float* bx = (g == 0) ? bgx : (g == 1) ? bix : (g == 2) ? bfx : box;
        const float* bh = (g == 0) ? bgh : (g == 1) ? bih : (g == 2) ? bfh : boh;
        Wx4[col] = wx[j];
        bb4[col] = bx[j] + bh[j];
    }
}

__global__ void lstm_init(__bf16* __restrict__ h0, float* __restrict__ c0) {
    int idx = blockIdx.x * blockDim.x + threadIdx.x;
    if (idx >= BD * HD) return;
    h0[idx] = (__bf16)0.0f;
    c0[idx] = 0.0f;
}

// ---------------------------------------------------------------------------
// One LSTM timestep.
// Grid (4,32): WG tile = 128 rows (M) x 16 hidden cols (N), all 4 gates.
// 8 waves; wave w owns m-strip w*16. All waves share one B tile per k-step:
//   B tile = 4 gates x (K=32 x N=16) bf16 = 4KB, double-buffered in LDS,
//   staged with global_load_async_to_lds_b128 (1 instr/wave/k-step),
//   consumed via ds_load in WMMA-B layout (lane = N col, elements = K rows).
// ---------------------------------------------------------------------------
__global__ __launch_bounds__(256) void lstm_step(
    const __bf16* __restrict__ h_in, __bf16* __restrict__ h_out,
    float* __restrict__ c, const __bf16* __restrict__ Whbf,
    const float* __restrict__ Wx4, const float* __restrict__ bb4,
    const float* __restrict__ x, int t)
{
    __shared__ __align__(128) unsigned char smem[2 * 4096];

    const int lane = threadIdx.x & 31;
    const int wave = threadIdx.x >> 5;
    const int m0 = blockIdx.x * 128;
    const int n0 = blockIdx.y * 16;
    const int mstrip = m0 + wave * 16;

    // Async-stage assignment: wave -> (gate, 16B chunk of the 32B K-row slice)
    const int sgate  = wave >> 1;
    const int schunk = wave & 1;

    // Global source for this wave's async copy: lane = K row (stride G4).
    const __bf16* gSrc = Whbf + (size_t)lane * G4 + n0 + sgate * HD + schunk * 8;
    // LDS destinations (two buffers). Layout inside a buffer:
    //   gate*1024 + chunk*512 + lane*16  (chunk == N-half of the v16bf frag)
    const unsigned ldsDst0 =
        (unsigned)(uintptr_t)(smem + sgate * 1024 + schunk * 512 + lane * 16);
    const unsigned ldsDst1 = ldsDst0 + 4096u;

    // A fragment addressing (16x32 bf16 ISA layout):
    //   lane l<16 : row M=l,    K in {0..7} U {16..23}; lanes>=16: K offset +8
    const int arow = mstrip + (lane & 15);
    const __bf16* aRow = h_in + arow * HD + ((lane >> 4) << 3);

    // LDS read base for this lane (lane = K row of the B fragment).
    const unsigned char* rdl = smem + lane * 16;

    v8f acc0 = {}, acc1 = {}, acc2 = {}, acc3 = {};

    // Prologue: stage k-step 0 into buffer 0; load A frag 0 into registers.
    asm volatile("global_load_async_to_lds_b128 %0, %1, off"
                 :: "v"(ldsDst0), "v"(gSrc) : "memory");
    Frag16 a;
    a.h[0] = *(const v8bf*)(aRow);
    a.h[1] = *(const v8bf*)(aRow + 16);

    #pragma unroll 2
    for (int ks = 0; ks < 16; ++ks) {
        // Stage next B tile into the other buffer (wrap harmlessly on ks=15;
        // the stray tile is never read and S_ENDPGM waits for idle).
        {
            const int kn = (ks + 1) & 15;
            const __bf16* gNext = gSrc + (size_t)(kn * 32) * G4;
            const unsigned dst = (ks & 1) ? ldsDst0 : ldsDst1;
            asm volatile("global_load_async_to_lds_b128 %0, %1, off"
                         :: "v"(dst), "v"(gNext) : "memory");
        }
        // Wait for OWN async copy of k-step ks (in-order; next one may fly).
        asm volatile("s_wait_asynccnt 0x1" ::: "memory");
        __syncthreads();   // all waves' B(ks) now visible in LDS

        // Prefetch next A fragment (registers) while WMMAs run.
        const int kn = (ks + 1) & 15;
        Frag16 an;
        an.h[0] = *(const v8bf*)(aRow + kn * 32);
        an.h[1] = *(const v8bf*)(aRow + kn * 32 + 16);

        // Pull the 4 gate B fragments from LDS (lane = N col, elements = K in table order).
        const unsigned char* rb = smem + ((ks & 1) ? 4096u : 0u) + ((lane & 15) >> 3) * 512 + (lane & 7) * 2;
        Frag16 b0, b1, b2, b3;
        #pragma unroll
        for (int s = 0; s < 16; ++s) {
            const int kk = ((lane >> 4) << 3) + ((s < 8) ? s : (s + 8));
            const unsigned char* p = rb + kk * 16;
            b0.v[s] = *(const __bf16*)(p);
            b1.v[s] = *(const __bf16*)(p + 1024);
            b2.v[s] = *(const __bf16*)(p + 2048);
            b3.v[s] = *(const __bf16*)(p + 3072);
        }

        acc0 = __builtin_amdgcn_wmma_f32_16x16x32_bf16(false, a.v, false, b0.v, (short)0, acc0, false, false);
        acc1 = __builtin_amdgcn_wmma_f32_16x16x32_bf16(false, a.v, false, b1.v, (short)0, acc1, false, false);
        acc2 = __builtin_amdgcn_wmma_f32_16x16x32_bf16(false, a.v, false, b2.v, (short)0, acc2, false, false);
        acc3 = __builtin_amdgcn_wmma_f32_16x16x32_bf16(false, a.v, false, b3.v, (short)0, acc3, false, false);

        a = an;
        __syncthreads();   // everyone done reading this buffer before restage
    }

    // Epilogue. C/D layout: VGPR r, lane l -> M = r + (l>=16)*8, N = l&15.
    const int rowbase = mstrip + (lane >> 4) * 8;
    const int j = n0 + (lane & 15);

    float xv[8];
    #pragma unroll
    for (int r = 0; r < 8; ++r) xv[r] = x[(rowbase + r) * SD + t];

    const float wx0 = Wx4[j],          bv0 = bb4[j];
    const float wx1 = Wx4[HD + j],     bv1 = bb4[HD + j];
    const float wx2 = Wx4[2 * HD + j], bv2 = bb4[2 * HD + j];
    const float wx3 = Wx4[3 * HD + j], bv3 = bb4[3 * HD + j];

    #pragma unroll
    for (int r = 0; r < 8; ++r) {
        const int b = rowbase + r;
        const float xb = xv[r];
        const float Gg = acc0[r] + xb * wx0 + bv0;
        const float Gi = acc1[r] + xb * wx1 + bv1;
        const float Gf = acc2[r] + xb * wx2 + bv2;
        const float Go = acc3[r] + xb * wx3 + bv3;
        const float gg = tanhfast(Gg);
        const float ii = sigf(Gi);
        const float ff = sigf(Gf);
        const float oo = sigf(Go);
        const int off = b * HD + j;
        const float cn = gg * ii + c[off] * ff;
        c[off] = cn;
        h_out[off] = (__bf16)(tanhfast(cn) * oo);
    }
}

// ---------------------------------------------------------------------------
// Output projection: out(512x128) = h(bf16) @ Wout(512x128) + bout. Tiny GEMM.
// ---------------------------------------------------------------------------
__global__ void lstm_out(const __bf16* __restrict__ h, const float* __restrict__ Wout,
                         const float* __restrict__ bout, float* __restrict__ out) {
    int idx = blockIdx.x * blockDim.x + threadIdx.x;
    if (idx >= BD * OD) return;
    const int b = idx >> 7;
    const int o = idx & (OD - 1);
    float s = bout[o];
    const __bf16* hr = h + b * HD;
    #pragma unroll 8
    for (int j = 0; j < HD; ++j) s += (float)hr[j] * Wout[j * OD + o];
    out[idx] = s;
}

extern "C" void kernel_launch(void* const* d_in, const int* in_sizes, int n_in,
                              void* d_out, int out_size, void* d_ws, size_t ws_size,
                              hipStream_t stream) {
    const float* x    = (const float*)d_in[0];
    const float* Wgx  = (const float*)d_in[1];
    const float* bgx  = (const float*)d_in[2];
    const float* Wgh  = (const float*)d_in[3];
    const float* bgh  = (const float*)d_in[4];
    const float* Wix  = (const float*)d_in[5];
    const float* bix  = (const float*)d_in[6];
    const float* Wih  = (const float*)d_in[7];
    const float* bih  = (const float*)d_in[8];
    const float* Wfx  = (const float*)d_in[9];
    const float* bfx  = (const float*)d_in[10];
    const float* Wfh  = (const float*)d_in[11];
    const float* bfh  = (const float*)d_in[12];
    const float* Wox  = (const float*)d_in[13];
    const float* box_ = (const float*)d_in[14];
    const float* Woh  = (const float*)d_in[15];
    const float* boh  = (const float*)d_in[16];
    const float* Wout = (const float*)d_in[17];
    const float* bout = (const float*)d_in[18];

    // Workspace carve-up:
    //   Whbf: 2 MiB | Wx4: 8 KiB | bb4: 8 KiB | h ping/pong: 2x512 KiB | c: 1 MiB
    char* ws = (char*)d_ws;
    __bf16* Whbf  = (__bf16*)ws;
    float*  Wx4   = (float*)(ws + (size_t)(2u << 20));
    float*  bb4   = (float*)(ws + (size_t)(2u << 20) + 8192);
    __bf16* hping = (__bf16*)(ws + (size_t)(2u << 20) + 16384);
    __bf16* hpong = hping + BD * HD;
    float*  c     = (float*)(ws + (size_t)(2u << 20) + 16384 +
                             (size_t)2 * BD * HD * sizeof(__bf16));

    lstm_prep<<<(HD * G4 + 255) / 256, 256, 0, stream>>>(
        Wgh, Wih, Wfh, Woh, Wgx, Wix, Wfx, Wox,
        bgx, bix, bfx, box_, bgh, bih, bfh, boh, Whbf, Wx4, bb4);
    lstm_init<<<(BD * HD + 255) / 256, 256, 0, stream>>>(hping, c);

    __bf16* hin = hping;
    __bf16* hout = hpong;
    for (int t = 0; t < SD; ++t) {
        lstm_step<<<dim3(4, 32), 256, 0, stream>>>(hin, hout, c, Whbf, Wx4, bb4, x, t);
        __bf16* tmp = hin; hin = hout; hout = tmp;
    }

    lstm_out<<<(BD * OD + 255) / 256, 256, 0, stream>>>(hin, Wout, bout, (float*)d_out);
}
